// local_pointnet_5669356833335
// MI455X (gfx1250) — compile-verified, matrix operand order checked by reading, not run on hardware
//
#include <hip/hip_runtime.h>
#include <hip/hip_bf16.h>

typedef __attribute__((ext_vector_type(16))) __bf16       v16bf;
typedef __attribute__((ext_vector_type(8)))  float        v8f;
typedef __attribute__((ext_vector_type(8)))  unsigned int v8u;
typedef __attribute__((ext_vector_type(4)))  unsigned int u32x4;
typedef __attribute__((ext_vector_type(8)))  int          i32x8;
typedef __attribute__((ext_vector_type(4)))  int          i32x4;

typedef unsigned int   u32;
typedef unsigned short u16;

// Tensor Data Mover path: probe-verified builtins, guarded so either toolchain
// (or the host pass) falls back to synchronous b128 staging.
#if defined(__HIP_DEVICE_COMPILE__) && __has_builtin(__builtin_amdgcn_tensor_load_to_lds) && __has_builtin(__builtin_amdgcn_s_wait_tensorcnt)
#define USE_TDM 1
#else
#define USE_TDM 0
#endif

// ---------------------------------------------------------------------------
// helpers
// ---------------------------------------------------------------------------
__device__ __forceinline__ u16 f2bf(float f) {
    u32 u = __float_as_uint(f);
    u32 r = u + 0x7fffu + ((u >> 16) & 1u);   // round-to-nearest-even
    return (u16)(r >> 16);
}
__device__ __forceinline__ float bf2f(u16 h) {
    return __uint_as_float(((u32)h) << 16);
}
__device__ __forceinline__ float lrelu(float x) { return x > 0.f ? x : 0.01f * x; }

// LDS tile strides (u32 units). 20 and 68 keep 16-row fragment reads
// bank-conflict-free (20n, 68n mod 64 distinct for n=0..15) and rows 16B aligned.
#define AS_ST 20
#define BS_ST 20
#define H1_ST 68

// bf16 16x16x32 fragment: k-pairs {0..3}+4*hi and {8..11}+4*hi are contiguous
// u32 runs -> exactly two ds_load_b128 per fragment.
__device__ __forceinline__ v8u load_frag(const u32* row, int kt16, int hi) {
    const uint4* p = (const uint4*)(row + kt16 + (hi ? 4 : 0));
    uint4 lo = p[0];
    uint4 up = p[2];
    v8u r;
    r[0] = lo.x; r[1] = lo.y; r[2] = lo.z; r[3] = lo.w;
    r[4] = up.x; r[5] = up.y; r[6] = up.z; r[7] = up.w;
    return r;
}
__device__ __forceinline__ v8f wmma8(v8u a, v8u b, v8f acc) {
    return __builtin_amdgcn_wmma_f32_16x16x32_bf16(
        false, __builtin_bit_cast(v16bf, a),
        false, __builtin_bit_cast(v16bf, b),
        (short)0, acc, false, false);
}

#if USE_TDM
// TDM: async-copy a 128-row x 32-bf16 weight tile (row stride gStrideU32 u32)
// into LDS with 4-dword padding after every 16 dwords (our stride-20 layout).
__device__ __forceinline__ void tdm_load_wtile(const u32* gsrc, u32* lds_dst,
                                               int gStrideU32) {
    u32 lds = (u32)(uintptr_t)(void*)lds_dst;
    unsigned long long ga = (unsigned long long)(uintptr_t)gsrc;
    u32x4 g0;
    g0[0] = 1u;                                           // count=1, user D#
    g0[1] = lds;                                          // lds_addr
    g0[2] = (u32)ga;                                      // global_addr[31:0]
    g0[3] = ((u32)(ga >> 32) & 0x01FFFFFFu) | (2u << 30); // addr[56:32], type=2
    i32x8 g1;
    g1[0] = (int)((1u << 16) | (1u << 20) | (3u << 22) | (3u << 25));
            // data_size=2B, pad_enable, pad_interval=16dw, pad_amount=4dw
    g1[1] = (int)(32u << 16);                 // tensor_dim0 = 32 (low16 @ bit48)
    g1[2] = (int)(128u << 16);                // dim0 hi | tensor_dim1 = 128
    g1[3] = (int)(32u << 16);                 // dim1 hi | tile_dim0 = 32
    g1[4] = (int)128u;                        // tile_dim1 = 128, tile_dim2 = 0
    g1[5] = (int)(u32)(gStrideU32 * 2);       // tensor_dim0_stride (elements)
    g1[6] = (int)(128u << 16);                // stride0 hi | tensor_dim1_stride
    g1[7] = 0;
    i32x4 z4 = {0, 0, 0, 0};
#if defined(__clang_major__) && __clang_major__ >= 23
    i32x8 z8 = {0, 0, 0, 0, 0, 0, 0, 0};
    __builtin_amdgcn_tensor_load_to_lds(g0, g1, z4, z4, z8, 0);
#else
    __builtin_amdgcn_tensor_load_to_lds(g0, g1, z4, z4, 0);
#endif
}
#endif

// stage a 128x32 bf16 weight tile into Bs (TDM from wave 0, or sync b128)
__device__ __forceinline__ void stage_wtile(const u32* gsrc, u32* Bs,
                                            int gStrideU32, int tid, int wave) {
#if USE_TDM
    if (wave == 0) {
        tdm_load_wtile(gsrc, Bs, gStrideU32);
        __builtin_amdgcn_s_wait_tensorcnt(0);
    }
#else
    (void)wave;
#pragma unroll
    for (int s = 0; s < 2; ++s) {
        int q = tid + s * 256;              // 512 uint4 total
        int n = q >> 2, c4 = q & 3;
        *(uint4*)&Bs[n * BS_ST + c4 * 4] =
            ((const uint4*)gsrc)[(size_t)n * (gStrideU32 >> 2) + c4];
    }
#endif
}

// ---------------------------------------------------------------------------
// weight cast kernels: fp32 [K,N] -> bf16 transposed [N][Kpad] (zero padded)
// ---------------------------------------------------------------------------
__global__ void k_cast_lin_t(const float* __restrict__ w, u16* __restrict__ wt,
                             int K, int N, int Kpad) {
    int i = blockIdx.x * 256 + threadIdx.x;
    if (i >= N * Kpad) return;
    int n = i / Kpad, k = i % Kpad;
    float v = (k < K) ? w[(size_t)k * N + n] : 0.f;
    wt[i] = f2bf(v);
}

// conv weights [O=128, I=128, 27] -> bf16 [27][O][I]
__global__ void k_cast_conv_t(const float* __restrict__ w, u16* __restrict__ wt) {
    int i = blockIdx.x * 256 + threadIdx.x;
    if (i >= 27 * 128 * 128) return;
    int t = i / (128 * 128);
    int r = i % (128 * 128);
    int o = r / 128, ii = r % 128;
    wt[i] = f2bf(w[((size_t)o * 128 + ii) * 27 + t]);
}

__global__ void k_zero_u32(u32* p, unsigned n) {
    unsigned i = blockIdx.x * 256u + threadIdx.x;
    if (i < n) p[i] = 0u;
}

// ---------------------------------------------------------------------------
// Fused two-layer MLP block:
//   stage1: H1 = lrelu(A @ W1 + b1)   (A gathered/concat'ed or dense bf16)
//   stage2: Y  = A2 @ W2 + b2         (A2 = H1 kept in LDS)
//   epilogue: residual-add + lrelu, or 8-row max-pool + lrelu
// block = 256 threads (8 waves), tile = 64 rows x 128 cols
// ---------------------------------------------------------------------------
template <int KT1, bool GATHER, bool MAXPOOL, bool RESID>
__global__ __launch_bounds__(256) void k_fused_mlp(
        const u32* __restrict__ Au,      // !GATHER: bf16 [rowsIn][KT1*32]; GATHER: feat bf16 [*,128]
        const int* __restrict__ idx,     // GATHER only
        const float* __restrict__ xyzf,  // GATHER only: f32 [rowsIn,3]
        const u32* __restrict__ w1t,     // bf16 [128][KT1*32]
        const float* __restrict__ b1,
        const u32* __restrict__ w2t,     // bf16 [128][128]
        const float* __restrict__ b2,
        const float* __restrict__ residf,  // RESID only: f32 [rowsOut,128]
        float* __restrict__ outF,          // f32 [rowsOut,128]
        u16* __restrict__ outB,            // bf16 [rowsOut,128]
        int rowsIn, int rowsOut) {
    __shared__ __align__(16) u32 As[64 * AS_ST];
    __shared__ __align__(16) u32 Bs[128 * BS_ST];
    __shared__ __align__(16) u32 H1[64 * H1_ST];

    const int tid = threadIdx.x, lane = tid & 31, wave = tid >> 5;
    const int wr = wave & 3, wc = wave >> 2, hi = (lane >= 16), l15 = lane & 15;
    const int m0 = blockIdx.x * 64;

    v8f acc[4] = {};
    for (int kt = 0; kt < KT1; ++kt) {
        // ---- stage A tile: 256 uint4, one b128 per thread ----
        {
            int i = tid >> 2, c4 = tid & 3, r = m0 + i;
            uint4 v = {0, 0, 0, 0};
            if constexpr (!GATHER) {
                if (r < rowsIn)
                    v = ((const uint4*)Au)[(size_t)r * (KT1 * 4) + kt * 4 + c4];
            } else {
                if (kt < KT1 - 1) {               // gathered feature k-tiles
                    if (r < rowsIn) {
                        int g = idx[r];
                        v = ((const uint4*)Au)[(size_t)g * 16 + kt * 4 + c4];
                    }
                } else {                          // xyz (3 vals) + zero pad
                    if (r < rowsIn && c4 == 0) {
                        v.x = (u32)f2bf(xyzf[(size_t)r * 3 + 0]) |
                              ((u32)f2bf(xyzf[(size_t)r * 3 + 1]) << 16);
                        v.y = (u32)f2bf(xyzf[(size_t)r * 3 + 2]);
                    }
                }
            }
            *(uint4*)&As[i * AS_ST + c4 * 4] = v;
        }
        stage_wtile(w1t + kt * 16, Bs, KT1 * 16, tid, wave);
        __syncthreads();
        v8u af = load_frag(&As[(wr * 16 + l15) * AS_ST], 0, hi);
#pragma unroll
        for (int t = 0; t < 4; ++t) {
            v8u bf_ = load_frag(&Bs[(wc * 64 + t * 16 + l15) * BS_ST], 0, hi);
            acc[t] = wmma8(af, bf_, acc[t]);
        }
        __syncthreads();
    }

    // stage1 epilogue: bias + lrelu -> H1 (bf16 in LDS)
#pragma unroll
    for (int t = 0; t < 4; ++t) {
        int col = wc * 64 + t * 16 + l15;
        float bb = b1[col];
#pragma unroll
        for (int j = 0; j < 8; ++j) {
            int rl = wr * 16 + (hi ? j + 8 : j);
            ((u16*)&H1[rl * H1_ST])[col] = f2bf(lrelu(acc[t][j] + bb));
        }
    }
    __syncthreads();

    // ---- stage 2: H1 @ W2 ----
    v8f acc2[4] = {};
    for (int kt = 0; kt < 4; ++kt) {
        stage_wtile(w2t + kt * 16, Bs, 64, tid, wave);
        __syncthreads();
        v8u af = load_frag(&H1[(wr * 16 + l15) * H1_ST], kt * 16, hi);
#pragma unroll
        for (int t = 0; t < 4; ++t) {
            v8u bf_ = load_frag(&Bs[(wc * 64 + t * 16 + l15) * BS_ST], 0, hi);
            acc2[t] = wmma8(af, bf_, acc2[t]);
        }
        __syncthreads();
    }

    // ---- epilogue ----
    if constexpr (MAXPOOL) {
        // 16-row tile = 2 groups of 8 rows; each lane owns exactly one group
#pragma unroll
        for (int t = 0; t < 4; ++t) {
            float v = acc2[t][0];
#pragma unroll
            for (int j = 1; j < 8; ++j) v = fmaxf(v, acc2[t][j]);
            int col = wc * 64 + t * 16 + l15;
            v = lrelu(v + b2[col]);
            int orow = (m0 >> 3) + wr * 2 + hi;
            if (orow < rowsOut) {
                outF[(size_t)orow * 128 + col] = v;
                outB[(size_t)orow * 128 + col] = f2bf(v);
            }
        }
    } else {
#pragma unroll
        for (int t = 0; t < 4; ++t) {
            int col = wc * 64 + t * 16 + l15;
            float bb = b2[col];
#pragma unroll
            for (int j = 0; j < 8; ++j) {
                int row = m0 + wr * 16 + (hi ? j + 8 : j);
                if (row < rowsOut) {
                    float v = acc2[t][j] + bb;
                    if constexpr (RESID) v += residf[(size_t)row * 128 + col];
                    v = lrelu(v);
                    outF[(size_t)row * 128 + col] = v;
                    outB[(size_t)row * 128 + col] = f2bf(v);
                }
            }
        }
    }
}

// ---------------------------------------------------------------------------
// plain GEMM + lrelu (fc4/fc5): Y = lrelu(A[rows,128] @ W + b) -> bf16
// ---------------------------------------------------------------------------
__global__ __launch_bounds__(256) void k_gemm_lrelu(
        const u32* __restrict__ Au, const u32* __restrict__ wt,
        const float* __restrict__ bias, u16* __restrict__ outB, int rows) {
    __shared__ __align__(16) u32 As[64 * AS_ST];
    __shared__ __align__(16) u32 Bs[128 * BS_ST];
    const int tid = threadIdx.x, lane = tid & 31, wave = tid >> 5;
    const int wr = wave & 3, wc = wave >> 2, hi = (lane >= 16), l15 = lane & 15;
    const int m0 = blockIdx.x * 64;

    v8f acc[4] = {};
    for (int kt = 0; kt < 4; ++kt) {
        {
            int i = tid >> 2, c4 = tid & 3, r = m0 + i;
            uint4 v = {0, 0, 0, 0};
            if (r < rows) v = ((const uint4*)Au)[(size_t)r * 16 + kt * 4 + c4];
            *(uint4*)&As[i * AS_ST + c4 * 4] = v;
        }
        stage_wtile(wt + kt * 16, Bs, 64, tid, wave);
        __syncthreads();
        v8u af = load_frag(&As[(wr * 16 + l15) * AS_ST], 0, hi);
#pragma unroll
        for (int t = 0; t < 4; ++t) {
            v8u bf_ = load_frag(&Bs[(wc * 64 + t * 16 + l15) * BS_ST], 0, hi);
            acc[t] = wmma8(af, bf_, acc[t]);
        }
        __syncthreads();
    }
#pragma unroll
    for (int t = 0; t < 4; ++t) {
        int col = wc * 64 + t * 16 + l15;
        float bb = bias[col];
#pragma unroll
        for (int j = 0; j < 8; ++j) {
            int row = m0 + wr * 16 + (hi ? j + 8 : j);
            if (row < rows) outB[(size_t)row * 128 + col] = f2bf(lrelu(acc[t][j] + bb));
        }
    }
}

// ---------------------------------------------------------------------------
// 3x3x3 conv, 128->128 ch over 64^3 grid as implicit GEMM.
// block = full z-column (64 voxels) x 128 out channels; each weight tile is
// reused by 4 WMMAs per wave (b-fragment kept in registers).
// ---------------------------------------------------------------------------
template <bool ACT>
__global__ __launch_bounds__(256) void k_conv3d(
        const u32* __restrict__ gin,   // bf16 grid [64][64][64][128] as u32
        const u32* __restrict__ wt,    // bf16 [27][128(o)][128(i)] as u32
        const float* __restrict__ bias,
        u16* __restrict__ gout) {
    __shared__ __align__(16) u32 As[64 * AS_ST];
    __shared__ __align__(16) u32 Bs[128 * BS_ST];
    const int tid = threadIdx.x, lane = tid & 31, wave = tid >> 5;
    const int hi = (lane >= 16), l15 = lane & 15;
    const int y = blockIdx.x, x = blockIdx.y;

    v8f acc[4] = {};
    for (int t = 0; t < 27; ++t) {
        int dx = t / 9 - 1, dy = (t / 3) % 3 - 1, dz = t % 3 - 1;
        int xx = x + dx, yy = y + dy;
        bool pok = (xx >= 0 && xx < 64 && yy >= 0 && yy < 64);
        for (int kt = 0; kt < 4; ++kt) {
            {   // A tile: 64 z-rows x 4 uint4, one b128 per thread
                int i = tid >> 2, c4 = tid & 3;
                int z = i + dz;
                uint4 v = {0, 0, 0, 0};
                if (pok && z >= 0 && z < 64)
                    v = ((const uint4*)gin)[(((size_t)xx * 64 + yy) * 64 + z) * 16 + kt * 4 + c4];
                *(uint4*)&As[i * AS_ST + c4 * 4] = v;
            }
            stage_wtile(wt + ((size_t)t * 128) * 64 + kt * 16, Bs, 64, tid, wave);
            __syncthreads();
            v8u bf_ = load_frag(&Bs[(wave * 16 + l15) * BS_ST], 0, hi);
#pragma unroll
            for (int mt = 0; mt < 4; ++mt) {
                v8u af = load_frag(&As[(mt * 16 + l15) * AS_ST], 0, hi);
                acc[mt] = wmma8(af, bf_, acc[mt]);
            }
            __syncthreads();
        }
    }
    int col = wave * 16 + l15;
    float bb = bias[col];
#pragma unroll
    for (int mt = 0; mt < 4; ++mt) {
#pragma unroll
        for (int j = 0; j < 8; ++j) {
            int z = mt * 16 + (hi ? j + 8 : j);
            float v = acc[mt][j] + bb;
            if (ACT) v = lrelu(v);
            gout[((((size_t)x * 64 + y) * 64 + z) * 128) + col] = f2bf(v);
        }
    }
}

// ---------------------------------------------------------------------------
// scatter voxel features into grid / gather back (+lrelu) / output heads
// ---------------------------------------------------------------------------
__global__ void k_scatter(const u32* __restrict__ src, const int* __restrict__ vi,
                          u32* __restrict__ grid, int M) {
    int i = blockIdx.x * 256 + threadIdx.x;
    if (i >= M * 64) return;
    int m = i >> 6, q = i & 63;
    size_t vl = (((size_t)vi[m * 3] * 64 + vi[m * 3 + 1]) * 64 + vi[m * 3 + 2]);
    grid[vl * 64 + q] = src[(size_t)m * 64 + q];
}

__global__ void k_gather_act(const u16* __restrict__ grid, const int* __restrict__ vi,
                             u16* __restrict__ out, int M) {
    int i = blockIdx.x * 256 + threadIdx.x;
    if (i >= M * 128) return;
    int m = i >> 7, c = i & 127;
    size_t vl = (((size_t)vi[m * 3] * 64 + vi[m * 3 + 1]) * 64 + vi[m * 3 + 2]);
    out[(size_t)m * 128 + c] = f2bf(lrelu(bf2f(grid[vl * 128 + c])));
}

__global__ void k_heads(const u16* __restrict__ h, const float* __restrict__ wob,
                        const float* __restrict__ bob, const float* __restrict__ wof,
                        const float* __restrict__ bof, float* __restrict__ out, int M) {
    int i = blockIdx.x * 256 + threadIdx.x;
    if (i >= M * 6) return;
    int m = i / 6, j = i % 6;
    const u16* hr = h + (size_t)m * 128;
    const float* w = (j < 3) ? wob : wof;
    int jj = (j < 3) ? j : j - 3;
    float s = (j < 3) ? bob[jj] : bof[jj];
    for (int k = 0; k < 128; ++k) s += bf2f(hr[k]) * w[k * 3 + jj];
    if (j < 3) out[(size_t)m * 3 + jj] = 1.f / (1.f + __expf(-s));
    else out[(size_t)M * 3 + (size_t)m * 3 + jj] = s;
}

// ---------------------------------------------------------------------------
// host launcher
// ---------------------------------------------------------------------------
static inline void wbpair(void* const* d_in, const int* sz, int i,
                          const float** w, const float** b) {
    if (sz[i] >= sz[i + 1]) { *w = (const float*)d_in[i]; *b = (const float*)d_in[i + 1]; }
    else                    { *w = (const float*)d_in[i + 1]; *b = (const float*)d_in[i]; }
}

extern "C" void kernel_launch(void* const* d_in, const int* in_sizes, int n_in,
                              void* d_out, int out_size, void* d_ws, size_t ws_size,
                              hipStream_t stream) {
    (void)n_in; (void)out_size; (void)ws_size;

    const int*   pc_idx = (const int*)d_in[0];
    const float* pc_xyz = (const float*)d_in[1];
    const int*   vi     = (const int*)d_in[2];
    const int*   vx_idx = (const int*)d_in[3];
    const float* vx_xyz = (const float*)d_in[4];

    const int rows_pc = in_sizes[0];      // N*8 = 524288
    const int N       = rows_pc / 8;      // 65536
    const int rows_vx = in_sizes[3];      // M*8 = 800000
    const int M       = in_sizes[2] / 3;  // 100000

    // ---- params (setup_inputs insertion order; w/b disambiguated by size) ----
    int p = 5;
    const float *c0l1w, *c0l1b, *c0l2w, *c0l2b;
    wbpair(d_in, in_sizes, p, &c0l1w, &c0l1b); p += 2;
    wbpair(d_in, in_sizes, p, &c0l2w, &c0l2b); p += 2;
    const float *rl1w[7], *rl1b[7], *rl2w[7], *rl2b[7];
    for (int i = 0; i < 7; ++i) {
        wbpair(d_in, in_sizes, p, &rl1w[i], &rl1b[i]); p += 2;
        wbpair(d_in, in_sizes, p, &rl2w[i], &rl2b[i]); p += 2;
    }
    const float *cl1w[6], *cl1b[6], *cl2w[6], *cl2b[6];
    for (int i = 0; i < 6; ++i) {
        wbpair(d_in, in_sizes, p, &cl1w[i], &cl1b[i]); p += 2;
        wbpair(d_in, in_sizes, p, &cl2w[i], &cl2b[i]); p += 2;
    }
    const float *kw[3], *kb[3];
    for (int i = 0; i < 3; ++i) { wbpair(d_in, in_sizes, p, &kw[i], &kb[i]); p += 2; }
    const float *fc4w, *fc4b, *fc5w, *fc5b, *obw, *obb, *ofw, *ofb;
    wbpair(d_in, in_sizes, p, &fc4w, &fc4b); p += 2;
    wbpair(d_in, in_sizes, p, &fc5w, &fc5b); p += 2;
    wbpair(d_in, in_sizes, p, &obw, &obb);   p += 2;
    wbpair(d_in, in_sizes, p, &ofw, &ofb);   p += 2;

    // ---- workspace layout ----
    char* wsb = (char*)d_ws;
    size_t off = 0;
    auto alloc = [&](size_t bytes) -> void* {
        void* q = wsb + off;
        off += (bytes + 255) & ~(size_t)255;
        return q;
    };
    u16* w_c0l1 = (u16*)alloc(128 * 32 * 2);
    u16* w_c0l2 = (u16*)alloc(128 * 128 * 2);
    u16 *w_r1[7], *w_r2[7];
    for (int i = 0; i < 7; ++i) { w_r1[i] = (u16*)alloc(128 * 128 * 2); w_r2[i] = (u16*)alloc(128 * 128 * 2); }
    u16 *w_c1[6], *w_c2[6];
    for (int i = 0; i < 6; ++i) { w_c1[i] = (u16*)alloc(128 * 160 * 2); w_c2[i] = (u16*)alloc(128 * 128 * 2); }
    u16* w_k[3];
    for (int i = 0; i < 3; ++i) w_k[i] = (u16*)alloc((size_t)27 * 128 * 128 * 2);
    u16* w_fc4 = (u16*)alloc(128 * 128 * 2);
    u16* w_fc5 = (u16*)alloc(128 * 128 * 2);
    float* bufA_f = (float*)alloc((size_t)M * 128 * 4);
    float* bufB_f = (float*)alloc((size_t)M * 128 * 4);
    u16* bufA_b = (u16*)alloc((size_t)M * 128 * 2);
    u16* bufB_b = (u16*)alloc((size_t)M * 128 * 2);
    u16* grid0 = (u16*)alloc((size_t)64 * 64 * 64 * 128 * 2);
    u16* grid1 = (u16*)alloc((size_t)64 * 64 * 64 * 128 * 2);

    auto cblk = [](int n) { return (n + 255) / 256; };

    // ---- cast weights to transposed/padded bf16 ----
    k_cast_lin_t<<<cblk(128 * 32), 256, 0, stream>>>(c0l1w, w_c0l1, 3, 128, 32);
    k_cast_lin_t<<<cblk(128 * 128), 256, 0, stream>>>(c0l2w, w_c0l2, 128, 128, 128);
    for (int i = 0; i < 7; ++i) {
        k_cast_lin_t<<<cblk(128 * 128), 256, 0, stream>>>(rl1w[i], w_r1[i], 128, 128, 128);
        k_cast_lin_t<<<cblk(128 * 128), 256, 0, stream>>>(rl2w[i], w_r2[i], 128, 128, 128);
    }
    for (int i = 0; i < 6; ++i) {
        k_cast_lin_t<<<cblk(128 * 160), 256, 0, stream>>>(cl1w[i], w_c1[i], 131, 128, 160);
        k_cast_lin_t<<<cblk(128 * 128), 256, 0, stream>>>(cl2w[i], w_c2[i], 128, 128, 128);
    }
    for (int i = 0; i < 3; ++i)
        k_cast_conv_t<<<cblk(27 * 128 * 128), 256, 0, stream>>>(kw[i], w_k[i]);
    k_cast_lin_t<<<cblk(128 * 128), 256, 0, stream>>>(fc4w, w_fc4, 128, 128, 128);
    k_cast_lin_t<<<cblk(128 * 128), 256, 0, stream>>>(fc5w, w_fc5, 128, 128, 128);

    const int blkPC = (rows_pc + 63) / 64;
    const int blkN  = (N + 63) / 64;
    const int blkVX = (rows_vx + 63) / 64;
    const int blkM  = (M + 63) / 64;

    // ---- c0 (xyz MLP + maxpool) -> bufA ----
    k_fused_mlp<1, true, true, false><<<blkPC, 256, 0, stream>>>(
        nullptr, pc_idx, pc_xyz, (const u32*)w_c0l1, c0l1b, (const u32*)w_c0l2, c0l2b,
        nullptr, bufA_f, bufA_b, rows_pc, N);
    // ---- r0 -> bufB ----
    k_fused_mlp<4, false, false, true><<<blkN, 256, 0, stream>>>(
        (const u32*)bufA_b, nullptr, nullptr, (const u32*)w_r1[0], rl1b[0],
        (const u32*)w_r2[0], rl2b[0], bufA_f, bufB_f, bufB_b, N, N);
    // ---- c[0..4] + r[1..5] ----
    for (int i = 0; i < 5; ++i) {
        k_fused_mlp<5, true, true, false><<<blkPC, 256, 0, stream>>>(
            (const u32*)bufB_b, pc_idx, pc_xyz, (const u32*)w_c1[i], cl1b[i],
            (const u32*)w_c2[i], cl2b[i], nullptr, bufA_f, bufA_b, rows_pc, N);
        k_fused_mlp<4, false, false, true><<<blkN, 256, 0, stream>>>(
            (const u32*)bufA_b, nullptr, nullptr, (const u32*)w_r1[i + 1], rl1b[i + 1],
            (const u32*)w_r2[i + 1], rl2b[i + 1], bufA_f, bufB_f, bufB_b, N, N);
    }
    // ---- c[5] (voxel gather) + r[6] ----
    k_fused_mlp<5, true, true, false><<<blkVX, 256, 0, stream>>>(
        (const u32*)bufB_b, vx_idx, vx_xyz, (const u32*)w_c1[5], cl1b[5],
        (const u32*)w_c2[5], cl2b[5], nullptr, bufA_f, bufA_b, rows_vx, M);
    k_fused_mlp<4, false, false, true><<<blkM, 256, 0, stream>>>(
        (const u32*)bufA_b, nullptr, nullptr, (const u32*)w_r1[6], rl1b[6],
        (const u32*)w_r2[6], rl2b[6], bufA_f, bufB_f, bufB_b, M, M);

    // ---- scatter into dense grid ----
    const unsigned gridU32 = 64u * 64u * 64u * 64u;  // 16.7M u32
    k_zero_u32<<<cblk((int)gridU32), 256, 0, stream>>>((u32*)grid0, gridU32);
    k_scatter<<<cblk(M * 64), 256, 0, stream>>>((const u32*)bufB_b, vi, (u32*)grid0, M);

    // ---- 3x3x3 convs (implicit GEMM over full z-columns) ----
    dim3 cgrid(64, 64);
    k_conv3d<true><<<cgrid, 256, 0, stream>>>((const u32*)grid0, (const u32*)w_k[0], kb[0], grid1);
    k_conv3d<true><<<cgrid, 256, 0, stream>>>((const u32*)grid1, (const u32*)w_k[1], kb[1], grid0);
    k_conv3d<false><<<cgrid, 256, 0, stream>>>((const u32*)grid0, (const u32*)w_k[2], kb[2], grid1);

    // ---- gather + lrelu, fc4, fc5, heads ----
    k_gather_act<<<cblk(M * 128), 256, 0, stream>>>(grid1, vi, bufA_b, M);
    k_gemm_lrelu<<<blkM, 256, 0, stream>>>((const u32*)bufA_b, (const u32*)w_fc4, fc4b, bufB_b, M);
    k_gemm_lrelu<<<blkM, 256, 0, stream>>>((const u32*)bufB_b, (const u32*)w_fc5, fc5b, bufA_b, M);
    k_heads<<<cblk(M * 6), 256, 0, stream>>>(bufA_b, obw, obb, ofw, ofb, (float*)d_out, M);
}
